// SpatialAttnProcessor2_0_20650202759555
// MI455X (gfx1250) — compile-verified, matrix operand order checked by reading, not run on hardware
//
#include <hip/hip_runtime.h>
#include <hip/hip_bf16.h>

// ---------------------------------------------------------------------------
// SDXL write-mode spatial attention, bf16-WMMA implementation for gfx1250.
//   q,k,v = hs @ W{q,k,v};  flash-attn with bool mask;  out = attn @ Wo + bo
// All matmuls via v_wmma_f32_16x16x32_bf16; fragments fed by ds_load_b128.
// ---------------------------------------------------------------------------

typedef __attribute__((ext_vector_type(16))) __bf16 v16bf;
typedef __attribute__((ext_vector_type(8)))  float  v8f;

#define SEQ   2304          // id_length * T
#define RTOT  4608          // 2 * SEQ
#define CDIM  1280
#define NHEAD 20
#define HDIM  64
#define NEG_BIG (-3.402823466e38f)

static __device__ __forceinline__ unsigned short f2bf_bits(float f){
  unsigned u = __builtin_bit_cast(unsigned, f);
  u += 0x7FFFu + ((u >> 16) & 1u);                 // round-to-nearest-even
  return (unsigned short)(u >> 16);
}
static __device__ __forceinline__ __bf16 f2bf(float f){
  return __builtin_bit_cast(__bf16, f2bf_bits(f));
}
static __device__ __forceinline__ v8f vzero8(){
  v8f z;
#pragma unroll
  for (int i = 0; i < 8; ++i) z[i] = 0.0f;
  return z;
}

struct U32x8 { uint4 lo, hi; };
union  BU    { uint4 u; __bf16 h[8]; };

// 16-bit A/B fragment (ISA 7.12.2 row pattern): lane&15 picks the row,
// lane>=16 adds +8 to the K sub-offset; contiguous -> two ds_load_b128.
static __device__ __forceinline__ v16bf frag_row(const __bf16* lds, int row0,
                                                 int k0, int ld, int lane){
  int r  = row0 + (lane & 15);
  int kh = (lane & 16) ? 8 : 0;
  const __bf16* p = lds + r * ld + k0 + kh;
  U32x8 t;
  t.lo = *(const uint4*)(p);        // K = k0+kh .. +7   -> VGPR 0..3
  t.hi = *(const uint4*)(p + 16);   // K = k0+kh+16..+23 -> VGPR 4..7
  return __builtin_bit_cast(v16bf, t);
}

// ---------------------------------------------------------------------------
// fp32 -> bf16 conversion, vectorized (float4 in, ushort4 out)
// ---------------------------------------------------------------------------
__global__ void __launch_bounds__(256)
cvt_f32_bf16(const float4* __restrict__ src, ushort4* __restrict__ dst, int n4){
  int i = blockIdx.x * blockDim.x + threadIdx.x;
  if (i >= n4) return;
  float4 f = src[i];
  ushort4 o;
  o.x = f2bf_bits(f.x); o.y = f2bf_bits(f.y);
  o.z = f2bf_bits(f.z); o.w = f2bf_bits(f.w);
  dst[i] = o;
}

// ---------------------------------------------------------------------------
// GEMM: C[r][n] = sum_k A[r][k] * B[k][n]   (A,B bf16; fp32 accumulate)
// Block tile 64(M) x 128(N), BK=32, 8 waves x 32x32 sub-tile.
// B tile stored TRANSPOSED in LDS so both fragment loads are ds_load_b128.
// Register-staged pipeline: next global tile loads overlap the WMMAs.
// ---------------------------------------------------------------------------
#define BM 64
#define BN 128
#define BK 32
#define LDA_S 48     // 32 + 16 pad -> 96B rows (16B-aligned vectors)
#define LDBT  48     // Bt[n][k] pitch

template<bool F32OUT>
__global__ void __launch_bounds__(256)
gemm_bf16_wmma(const unsigned short* __restrict__ Aus,
               const unsigned short* __restrict__ Bus,
               unsigned short* __restrict__ outBus,
               float* __restrict__ outF,
               const float* __restrict__ bias,
               float oscale, int K, int N){
  const __bf16* A = (const __bf16*)Aus;
  const __bf16* B = (const __bf16*)Bus;
  __bf16* outB = (__bf16*)outBus;

  __shared__ __bf16 As[BM * LDA_S];
  __shared__ __bf16 Bt[BN * LDBT];          // transposed: Bt[n][k]

  int tid  = threadIdx.x;
  int wave = tid >> 5;
  int lane = tid & 31;
  int r0 = blockIdx.y * BM;
  int n0 = blockIdx.x * BN;
  int wm = (wave & 1) * 32;                 // 2 waves along M
  int wn = (wave >> 1) * 32;                // 4 waves along N

  int arow = tid >> 2,  akc = (tid & 3) * 8;   // A: 64 rows x 32 k
  int brow = tid >> 4,  bnc = (tid & 15) * 8;  // B: 32 k x 128 n (2 halves)

  const __bf16* Ag  = A + (size_t)(r0 + arow) * K + akc;
  const __bf16* Bg0 = B + (size_t)brow * N + n0 + bnc;
  const __bf16* Bg1 = B + (size_t)(brow + 16) * N + n0 + bnc;

  uint4 aR = *(const uint4*)(Ag);
  BU b0R, b1R;
  b0R.u = *(const uint4*)(Bg0);
  b1R.u = *(const uint4*)(Bg1);

  v8f acc00 = vzero8(), acc01 = vzero8(), acc10 = vzero8(), acc11 = vzero8();

  for (int k0 = 0; k0 < K; k0 += BK){
    // commit staged registers (tile k0) to LDS; B transposed via b16 scatter
    *(uint4*)(&As[arow * LDA_S + akc]) = aR;
#pragma unroll
    for (int j = 0; j < 8; ++j){
      Bt[(bnc + j) * LDBT + brow]      = b0R.h[j];
      Bt[(bnc + j) * LDBT + brow + 16] = b1R.h[j];
    }
    __syncthreads();

    int kn = k0 + BK;
    if (kn < K){                           // issue next tile loads early
      aR    = *(const uint4*)(Ag + kn);
      b0R.u = *(const uint4*)(Bg0 + (size_t)kn * N);
      b1R.u = *(const uint4*)(Bg1 + (size_t)kn * N);
    }

    v16bf a0  = frag_row(As, wm,      0, LDA_S, lane);
    v16bf a1  = frag_row(As, wm + 16, 0, LDA_S, lane);
    v16bf bb0 = frag_row(Bt, wn,      0, LDBT,  lane);
    v16bf bb1 = frag_row(Bt, wn + 16, 0, LDBT,  lane);
    acc00 = __builtin_amdgcn_wmma_f32_16x16x32_bf16(false, a0, false, bb0, (short)0, acc00, false, false);
    acc01 = __builtin_amdgcn_wmma_f32_16x16x32_bf16(false, a0, false, bb1, (short)0, acc01, false, false);
    acc10 = __builtin_amdgcn_wmma_f32_16x16x32_bf16(false, a1, false, bb0, (short)0, acc10, false, false);
    acc11 = __builtin_amdgcn_wmma_f32_16x16x32_bf16(false, a1, false, bb1, (short)0, acc11, false, false);
    __syncthreads();
  }

  // epilogue: C layout m = v + 8*(lane>=16), n = lane&15  (branch-free)
  int hi = (lane & 16) ? 8 : 0;
  int cn = lane & 15;
#pragma unroll
  for (int fi = 0; fi < 2; ++fi){
#pragma unroll
    for (int fj = 0; fj < 2; ++fj){
      const v8f& a = fi == 0 ? (fj == 0 ? acc00 : acc01)
                             : (fj == 0 ? acc10 : acc11);
      int rr0 = r0 + wm + fi * 16 + hi;
      int cc  = n0 + wn + fj * 16 + cn;
      if constexpr (F32OUT){
        float bval = bias[cc];
#pragma unroll
        for (int v = 0; v < 8; ++v)
          outF[(size_t)(rr0 + v) * N + cc] = a[v] + bval;
      } else {
#pragma unroll
        for (int v = 0; v < 8; ++v)
          outB[(size_t)(rr0 + v) * N + cc] = f2bf(a[v] * oscale);
      }
    }
  }
}

// ---------------------------------------------------------------------------
// Flash attention: block = (batch b, head h, 64-row q-tile).
// Online softmax; V staged transposed so P@V B-fragments are ds_load_b128.
// 1/sqrt(64) is pre-folded into Q, mask staged through LDS.
// ---------------------------------------------------------------------------
#define QT  64
#define LDQ 72       // bf16 tiles: 64 + 8 pad (144B rows)

__global__ void __launch_bounds__(256)
flash_attn_wmma(const unsigned short* __restrict__ Qus,
                const unsigned short* __restrict__ Kus,
                const unsigned short* __restrict__ Vus,
                const unsigned char* __restrict__ mask,
                unsigned short* __restrict__ AOus){
  const __bf16* Qb = (const __bf16*)Qus;
  const __bf16* Kb = (const __bf16*)Kus;
  const __bf16* Vb = (const __bf16*)Vus;
  __bf16* AOb = (__bf16*)AOus;

  __shared__ __bf16 Qs [QT * LDQ];
  __shared__ __bf16 KPs[QT * LDQ];          // K tile, then reused as P tile
  __shared__ __bf16 Vt [QT * LDQ];          // V transposed: Vt[d][kt]
  __shared__ float  Sts [QT * QT];          // fp32 scores
  __shared__ float  Oacc[QT * QT];          // fp32 output accumulator
  __shared__ float  mrow[QT], lrow[QT], arow[QT];
  __shared__ unsigned char Ms[QT * QT];     // mask tile

  int tid  = threadIdx.x;
  int wave = tid >> 5;
  int lane = tid & 31;
  int q0 = blockIdx.x * QT;
  int h  = blockIdx.y;
  int b  = blockIdx.z;
  size_t base = (size_t)b * SEQ * CDIM + (size_t)h * HDIM;

  int srow = tid >> 3;                      // staging: 8 threads per row
  int scc  = (tid & 7) * 8;

  // load Q tile 64x64 (two 16B passes per thread)
#pragma unroll
  for (int p = 0; p < 2; ++p){
    int row = srow + p * 32;
    *(uint4*)(&Qs[row * LDQ + scc]) =
        *(const uint4*)(Qb + base + (size_t)(q0 + row) * CDIM + scc);
  }
  for (int i = tid; i < QT * QT; i += 256) Oacc[i] = 0.0f;
  if (tid < QT){ mrow[tid] = NEG_BIG; lrow[tid] = 0.0f; }
  __syncthreads();

  int hi = (lane & 16) ? 8 : 0;
  int cn = lane & 15;

  for (int kt0 = 0; kt0 < SEQ; kt0 += QT){
    // stage K (row-major), V (transposed), mask tile
#pragma unroll
    for (int p = 0; p < 2; ++p){
      int row = srow + p * 32;              // kt within tile
      *(uint4*)(&KPs[row * LDQ + scc]) =
          *(const uint4*)(Kb + base + (size_t)(kt0 + row) * CDIM + scc);
      BU v;
      v.u = *(const uint4*)(Vb + base + (size_t)(kt0 + row) * CDIM + scc);
#pragma unroll
      for (int j = 0; j < 8; ++j)
        Vt[(scc + j) * LDQ + row] = v.h[j];
    }
    {
      int mr = tid >> 2;
      int mc = (tid & 3) * 16;
      *(uint4*)(&Ms[mr * QT + mc]) =
          *(const uint4*)(mask + (size_t)(q0 + mr) * SEQ + kt0 + mc);
    }
    if (kt0 + QT < SEQ){                    // prefetch next K/V tiles
      const __bf16* kp = Kb + base + (size_t)(kt0 + QT + srow) * CDIM + scc;
      const __bf16* vp = Vb + base + (size_t)(kt0 + QT + srow) * CDIM + scc;
      __builtin_prefetch(kp, 0, 0);
      __builtin_prefetch(vp, 0, 0);
    }
    __syncthreads();

    // scores: 16 frags of 16x16, each wave owns 2;  K(=d)=64 -> 2 WMMAs
#pragma unroll
    for (int i = 0; i < 2; ++i){
      int f  = wave * 2 + i;
      int fm = (f >> 2) * 16;
      int fn = (f & 3) * 16;
      v8f acc = vzero8();
#pragma unroll
      for (int kk = 0; kk < HDIM; kk += 32){
        v16bf a  = frag_row(Qs,  fm, kk, LDQ, lane);    // Q[q][d]
        v16bf bt = frag_row(KPs, fn, kk, LDQ, lane);    // K^T: B[d][kt]=K[kt][d]
        acc = __builtin_amdgcn_wmma_f32_16x16x32_bf16(false, a, false, bt,
                                                      (short)0, acc, false, false);
      }
#pragma unroll
      for (int v = 0; v < 8; ++v){
        int m = fm + v + hi;
        int n = fn + cn;
        float s = acc[v];                                // scale folded into Q
        if (!Ms[m * QT + n]) s = NEG_BIG;
        Sts[m * QT + n] = s;
      }
    }
    __syncthreads();

    // online softmax, one thread per q row
    if (tid < QT){
      int r = tid;
      float mold = mrow[r];
      float tmax = NEG_BIG;
      for (int c = 0; c < QT; ++c) tmax = fmaxf(tmax, Sts[r * QT + c]);
      float mnew  = fmaxf(mold, tmax);
      float alpha = __expf(mold - mnew);
      float sum = 0.0f;
      for (int c = 0; c < QT; ++c){
        float p = __expf(Sts[r * QT + c] - mnew);
        sum += p;
        KPs[r * LDQ + c] = f2bf(p);                      // P tile (reuse K LDS)
      }
      lrow[r] = lrow[r] * alpha + sum;
      mrow[r] = mnew;
      arow[r] = alpha;
    }
    __syncthreads();

    // O = O*alpha + P @ V
#pragma unroll
    for (int i = 0; i < 2; ++i){
      int f  = wave * 2 + i;
      int fm = (f >> 2) * 16;
      int fn = (f & 3) * 16;
      v8f acc = vzero8();
#pragma unroll
      for (int kk = 0; kk < QT; kk += 32){
        v16bf a  = frag_row(KPs, fm, kk, LDQ, lane);     // P[q][kt]
        v16bf bb = frag_row(Vt,  fn, kk, LDQ, lane);     // V^T: B[kt][d]=Vt[d][kt]
        acc = __builtin_amdgcn_wmma_f32_16x16x32_bf16(false, a, false, bb,
                                                      (short)0, acc, false, false);
      }
#pragma unroll
      for (int v = 0; v < 8; ++v){
        int m = fm + v + hi;
        int n = fn + cn;
        Oacc[m * QT + n] = Oacc[m * QT + n] * arow[m] + acc[v];
      }
    }
    __syncthreads();
  }

  // finalize: divide by l, store bf16 attention output
  {
    int r  = tid >> 2;
    int c0 = (tid & 3) * 16;
    float inv = 1.0f / lrow[r];
#pragma unroll
    for (int c = c0; c < c0 + 16; ++c)
      AOb[base + (size_t)(q0 + r) * CDIM + c] = f2bf(Oacc[r * QT + c] * inv);
  }
}

// ---------------------------------------------------------------------------
// launcher
// ---------------------------------------------------------------------------
extern "C" void kernel_launch(void* const* d_in, const int* in_sizes, int n_in,
                              void* d_out, int out_size, void* d_ws, size_t ws_size,
                              hipStream_t stream){
  const float*         hs   = (const float*)d_in[0];
  const unsigned char* mask = (const unsigned char*)d_in[1];
  const float*         Wq   = (const float*)d_in[2];
  const float*         Wk   = (const float*)d_in[3];
  const float*         Wv   = (const float*)d_in[4];
  const float*         Wo   = (const float*)d_in[5];
  const float*         bo   = (const float*)d_in[6];
  float* out = (float*)d_out;

  char* w = (char*)d_ws;
  size_t off = 0;
  auto carve = [&](size_t bytes) -> char* {
    char* p = w + off;
    off = (off + bytes + 255) & ~(size_t)255;
    return p;
  };
  const size_t actB = (size_t)RTOT * CDIM * 2;   // bf16 activations
  const size_t wB   = (size_t)CDIM * CDIM * 2;   // bf16 weights
  unsigned short* hsb = (unsigned short*)carve(actB);
  unsigned short* Wqb = (unsigned short*)carve(wB);
  unsigned short* Wkb = (unsigned short*)carve(wB);
  unsigned short* Wvb = (unsigned short*)carve(wB);
  unsigned short* Wob = (unsigned short*)carve(wB);
  unsigned short* Qbf = (unsigned short*)carve(actB);
  unsigned short* Kbf = (unsigned short*)carve(actB);
  unsigned short* Vbf = (unsigned short*)carve(actB);
  unsigned short* AOb = (unsigned short*)carve(actB);

  const int nHS4 = RTOT * CDIM / 4;
  const int nW4  = CDIM * CDIM / 4;
  cvt_f32_bf16<<<(nHS4 + 255) / 256, 256, 0, stream>>>((const float4*)hs, (ushort4*)hsb, nHS4);
  cvt_f32_bf16<<<(nW4  + 255) / 256, 256, 0, stream>>>((const float4*)Wq, (ushort4*)Wqb, nW4);
  cvt_f32_bf16<<<(nW4  + 255) / 256, 256, 0, stream>>>((const float4*)Wk, (ushort4*)Wkb, nW4);
  cvt_f32_bf16<<<(nW4  + 255) / 256, 256, 0, stream>>>((const float4*)Wv, (ushort4*)Wvb, nW4);
  cvt_f32_bf16<<<(nW4  + 255) / 256, 256, 0, stream>>>((const float4*)Wo, (ushort4*)Wob, nW4);

  dim3 gg(CDIM / BN, RTOT / BM);                 // 10 x 72
  // fold 1/sqrt(HDIM) into Q so the flash kernel needs no per-score scale
  gemm_bf16_wmma<false><<<gg, 256, 0, stream>>>(hsb, Wqb, Qbf, nullptr, nullptr, 0.125f, CDIM, CDIM);
  gemm_bf16_wmma<false><<<gg, 256, 0, stream>>>(hsb, Wkb, Kbf, nullptr, nullptr, 1.0f,   CDIM, CDIM);
  gemm_bf16_wmma<false><<<gg, 256, 0, stream>>>(hsb, Wvb, Vbf, nullptr, nullptr, 1.0f,   CDIM, CDIM);

  dim3 ga(SEQ / QT, NHEAD, 2);                   // 36 x 20 x 2
  flash_attn_wmma<<<ga, 256, 0, stream>>>(Qbf, Kbf, Vbf, mask, AOb);

  gemm_bf16_wmma<true><<<gg, 256, 0, stream>>>(AOb, Wob, nullptr, out, bo, 1.0f, CDIM, CDIM);
}
